// myGNN_45655502356933
// MI455X (gfx1250) — matrix operand order repaired and checked by reading, not compiled, not run on hardware
//
#include <hip/hip_runtime.h>
#include <hip/hip_bf16.h>

// ---------------------------------------------------------------------------
// GATv2 x2 forward for MI455X (gfx1250, wave32).
// Dense projections: v_wmma_f32_16x16x32_bf16, fully unrolled K loop,
// pre-packed B fragments, 2 output tiles per wave (A-fragment reuse).
// Edge softmax/aggregation: irregular, L2-resident gather + f32 atomics.
// ---------------------------------------------------------------------------

#define D_IN   128
#define H1C1   256   // 4 heads * 64 ch
#define H2C2   128   // 1 head * 128 ch

typedef __attribute__((ext_vector_type(16))) __bf16        v16bf;
typedef __attribute__((ext_vector_type(8)))  float         v8f;
typedef __attribute__((ext_vector_type(8)))  unsigned int  v8u;

// ----- helpers -------------------------------------------------------------

__device__ __forceinline__ unsigned short f32_to_bf16_bits(float f) {
    unsigned int u = __float_as_uint(f);
    unsigned int r = 0x7FFFu + ((u >> 16) & 1u);   // round-to-nearest-even
    return (unsigned short)((u + r) >> 16);
}

// order-preserving float<->uint map so atomicMax(uint) == float max
__device__ __forceinline__ unsigned int ford(float f) {
    unsigned int b = __float_as_uint(f);
    return (b & 0x80000000u) ? ~b : (b | 0x80000000u);
}
__device__ __forceinline__ float ford_dec(unsigned int u) {
    unsigned int b = (u & 0x80000000u) ? (u & 0x7FFFFFFFu) : ~u;
    return __uint_as_float(b);
}

// A fragment: 16x32 bf16, row-major A, leading dim ldk (elements).
// lane 0-15: row = lane,      K = k0+0..15
// lane 16-31: row = lane-16,  K = k0+16..31
__device__ __forceinline__ v16bf load_a_frag(const unsigned short* A, int ldk,
                                             int row0, int k0, int lane) {
    int m  = row0 + (lane & 15);
    int kb = k0 + ((lane & 16) ? 16 : 0);
    const unsigned short* p = A + (size_t)m * ldk + kb;
    uint4 lo = *reinterpret_cast<const uint4*>(p);        // K = kb..kb+7
    uint4 hi = *reinterpret_cast<const uint4*>(p + 8);    // K = kb+8..kb+15
    v8u r;
    r[0]=lo.x; r[1]=lo.y; r[2]=lo.z; r[3]=lo.w;
    r[4]=hi.x; r[5]=hi.y; r[6]=hi.z; r[7]=hi.w;
    return __builtin_bit_cast(v16bf, r);
}

// Pre-packed B fragment: 32 bytes per lane, contiguous per (fragIdx, lane).
__device__ __forceinline__ v16bf load_b_packed(const unsigned short* Bp, int fi, int lane) {
    const uint4* p = reinterpret_cast<const uint4*>(Bp + ((size_t)fi * 32 + lane) * 16);
    uint4 lo = p[0], hi = p[1];
    v8u r;
    r[0]=lo.x; r[1]=lo.y; r[2]=lo.z; r[3]=lo.w;
    r[4]=hi.x; r[5]=hi.y; r[6]=hi.z; r[7]=hi.w;
    return __builtin_bit_cast(v16bf, r);
}

// Pack f32 W[K][N] (row-major) into bf16 WMMA B fragments.
// Fragment fi = tn*(K/32)+kk; lane layout uses the same contiguous K order as load_a_frag:
// lane 0-15: col = lane,     V_j = {K=kb+2j, kb+2j+1}, kb = kk*32
// lane 16-31: col = lane-16, kb = kk*32+16
__global__ void pack_b_kernel(const float* __restrict__ W,
                              unsigned short* __restrict__ packed, int N, int K) {
    int t = blockIdx.x * blockDim.x + threadIdx.x;
    int total = (N >> 4) * (K >> 5) * 32;
    if (t >= total) return;
    int lane = t & 31;
    int fi   = t >> 5;
    int kchunks = K >> 5;
    int kk = fi % kchunks;
    int tn = fi / kchunks;
    int n  = (tn << 4) + (lane & 15);
    int kb = (kk << 5) + ((lane & 16) ? 16 : 0);
    unsigned int r[8];
#pragma unroll
    for (int j = 0; j < 8; ++j) {
        unsigned int lo = f32_to_bf16_bits(W[(size_t)(kb + 2*j    ) * N + n]);
        unsigned int hi = f32_to_bf16_bits(W[(size_t)(kb + 2*j + 1) * N + n]);
        r[j] = lo | (hi << 16);
    }
    uint4* out = reinterpret_cast<uint4*>(packed + ((size_t)fi * 32 + lane) * 16);
    uint4 q0; q0.x=r[0]; q0.y=r[1]; q0.z=r[2]; q0.w=r[3];
    uint4 q1; q1.x=r[4]; q1.y=r[5]; q1.z=r[6]; q1.w=r[7];
    out[0] = q0; out[1] = q1;
}

// ----- BatchNorm -----------------------------------------------------------

__global__ void bn_stats_kernel(const float* __restrict__ x, float* __restrict__ sums, int N) {
    int f = threadIdx.x;                 // blockDim.x == 128
    float s = 0.f, s2 = 0.f;
    for (int r = blockIdx.x; r < N; r += gridDim.x) {
        float v = x[(size_t)r * D_IN + f];
        s += v; s2 += v * v;
    }
    atomicAdd(&sums[f], s);
    atomicAdd(&sums[D_IN + f], s2);
}

__global__ void bn_apply_kernel(const float* __restrict__ x, const float* __restrict__ sums,
                                const float* __restrict__ gamma, const float* __restrict__ beta,
                                unsigned short* __restrict__ hbf, int total, float invN) {
    int i = blockIdx.x * blockDim.x + threadIdx.x;
    if (i >= total) return;
    int f = i & (D_IN - 1);
    float mu  = sums[f] * invN;
    float var = sums[D_IN + f] * invN - mu * mu;
    float v = (x[i] - mu) * rsqrtf(var + 1e-5f) * gamma[f] + beta[f];
    hbf[i] = f32_to_bf16_bits(v);
}

// ----- WMMA GEMM: C[M,N] = A[M,K](bf16) @ B(packed bf16) + bias[N] ---------
// One wave computes two adjacent 16x16 output tiles (shared A fragment).
// K is compile-time -> fully unrolled chain of v_wmma ops.

template <int K>
__global__ void wmma_gemm_bias_kernel(const unsigned short* __restrict__ A,
                                      const unsigned short* __restrict__ Bp,
                                      const float* __restrict__ bias,
                                      float* __restrict__ C,
                                      int M, int N) {
    constexpr int KC = K / 32;
    int lane   = threadIdx.x & 31;
    int wave   = (blockIdx.x * blockDim.x + threadIdx.x) >> 5;
    int pairsN = N >> 5;                       // two 16-col tiles per wave
    if (wave >= (M >> 4) * pairsN) return;     // wave-uniform exit (EXEC stays full)
    int tm = wave / pairsN, tp = wave - tm * pairsN;
    int row0 = tm << 4;
    int col0 = tp << 5;
    int c0 = col0 + (lane & 15);
    int c1 = c0 + 16;

    float bv0 = bias[c0], bv1 = bias[c1];
    v8f acc0 = {bv0, bv0, bv0, bv0, bv0, bv0, bv0, bv0};
    v8f acc1 = {bv1, bv1, bv1, bv1, bv1, bv1, bv1, bv1};

    int fi0 = (2 * tp) * KC;       // fragment base for tile tn=2*tp
    int fi1 = fi0 + KC;            // and tn=2*tp+1
#pragma unroll
    for (int kk = 0; kk < KC; ++kk) {
        v16bf a  = load_a_frag(A, K, row0, kk * 32, lane);
        v16bf b0 = load_b_packed(Bp, fi0 + kk, lane);
        v16bf b1 = load_b_packed(Bp, fi1 + kk, lane);
        acc0 = __builtin_amdgcn_wmma_f32_16x16x32_bf16(
            false, a, false, b0, (short)0, acc0, false, false);
        acc1 = __builtin_amdgcn_wmma_f32_16x16x32_bf16(
            false, a, false, b1, (short)0, acc1, false, false);
    }

    int rbase = row0 + ((lane & 16) ? 8 : 0);
#pragma unroll
    for (int r = 0; r < 8; ++r) {
        float* rowp = C + (size_t)(rbase + r) * N;
        rowp[c0] = acc0[r];
        rowp[c1] = acc1[r];
    }
}

// ----- Edge phase ----------------------------------------------------------
// One wave per edge. H heads * C channels; 32/H lanes per head.

template <int H, int C>
__global__ void edge_alpha_kernel(const float* __restrict__ xl, const float* __restrict__ xr,
                                  const float* __restrict__ att,
                                  const int* __restrict__ srcp, const int* __restrict__ dstp,
                                  int E0, int Etot,
                                  float* __restrict__ alpha, unsigned int* __restrict__ amax) {
    constexpr int HC  = H * C;
    constexpr int FPL = HC / 32;    // features per lane
    constexpr int G   = 32 / H;     // lanes per head
    int wave = (blockIdx.x * blockDim.x + threadIdx.x) >> 5;
    int lane = threadIdx.x & 31;
    if (wave >= Etot) return;
    int s, d;
    if (wave < E0) { s = srcp[wave]; d = dstp[wave]; } else { s = wave - E0; d = s; }

    const float* pl = xl + (size_t)s * HC + lane * FPL;
    const float* pr = xr + (size_t)d * HC + lane * FPL;
    const float* pa = att + lane * FPL;
    float acc = 0.f;
#pragma unroll
    for (int i = 0; i < FPL; ++i) {
        float v = pl[i] + pr[i];
        v = (v > 0.f) ? v : 0.2f * v;           // leaky_relu(., 0.2)
        acc += v * pa[i];
    }
#pragma unroll
    for (int off = 1; off < G; off <<= 1) acc += __shfl_xor(acc, off, 32);
    if ((lane & (G - 1)) == 0) {
        int h = lane / G;
        alpha[(size_t)wave * H + h] = acc;
        atomicMax(&amax[(size_t)d * H + h], ford(acc));
    }
}

template <int H>
__global__ void edge_exp_kernel(float* __restrict__ alpha,
                                const unsigned int* __restrict__ amax,
                                float* __restrict__ denom,
                                const int* __restrict__ dstp, int E0, int Etot) {
    int i = blockIdx.x * blockDim.x + threadIdx.x;
    if (i >= Etot * H) return;
    int e = i / H, h = i - e * H;
    int d = (e < E0) ? dstp[e] : (e - E0);
    float m  = ford_dec(amax[(size_t)d * H + h]);
    float ea = __expf(alpha[i] - m);
    alpha[i] = ea;
    atomicAdd(&denom[(size_t)d * H + h], ea);
}

template <int H, int C>
__global__ void edge_agg_kernel(const float* __restrict__ xl, const float* __restrict__ alpha,
                                const float* __restrict__ denom,
                                const int* __restrict__ srcp, const int* __restrict__ dstp,
                                int E0, int Etot, float* __restrict__ agg) {
    constexpr int HC  = H * C;
    constexpr int FPL = HC / 32;
    constexpr int G   = 32 / H;
    int wave = (blockIdx.x * blockDim.x + threadIdx.x) >> 5;
    int lane = threadIdx.x & 31;
    if (wave >= Etot) return;
    int s, d;
    if (wave < E0) { s = srcp[wave]; d = dstp[wave]; } else { s = wave - E0; d = s; }
    int h = lane / G;
    float wgt = alpha[(size_t)wave * H + h] / denom[(size_t)d * H + h];
    const float* pl = xl  + (size_t)s * HC + lane * FPL;
    float*       po = agg + (size_t)d * HC + lane * FPL;
#pragma unroll
    for (int i = 0; i < FPL; ++i) atomicAdd(&po[i], pl[i] * wgt);
}

// ----- finalize ------------------------------------------------------------

__global__ void bias_lrelu_to_bf16_kernel(const float* __restrict__ agg,
                                          const float* __restrict__ bias,
                                          unsigned short* __restrict__ out,
                                          int total, int F) {
    int i = blockIdx.x * blockDim.x + threadIdx.x;
    if (i >= total) return;
    float v = agg[i] + bias[i % F];
    v = (v > 0.f) ? v : 0.01f * v;              // inter-layer leaky_relu(0.01)
    out[i] = f32_to_bf16_bits(v);
}

__global__ void add_bias_out_kernel(const float* __restrict__ agg,
                                    const float* __restrict__ bias,
                                    float* __restrict__ out, int total, int F) {
    int i = blockIdx.x * blockDim.x + threadIdx.x;
    if (i >= total) return;
    out[i] = agg[i] + bias[i % F];
}

// ---------------------------------------------------------------------------

static inline int ceil_div(long long a, long long b) { return (int)((a + b - 1) / b); }

extern "C" void kernel_launch(void* const* d_in, const int* in_sizes, int n_in,
                              void* d_out, int out_size, void* d_ws, size_t ws_size,
                              hipStream_t stream) {
    (void)n_in; (void)out_size; (void)ws_size;

    const float* x     = (const float*)d_in[0];
    const int*   eidx  = (const int*)d_in[1];   // JAX x64-off: int32
    const float* gamma = (const float*)d_in[2];
    const float* beta  = (const float*)d_in[3];
    const float* W1l   = (const float*)d_in[4];
    const float* b1l   = (const float*)d_in[5];
    const float* W1r   = (const float*)d_in[6];
    const float* b1r   = (const float*)d_in[7];
    const float* att1  = (const float*)d_in[8];
    const float* bias1 = (const float*)d_in[9];
    const float* W2l   = (const float*)d_in[10];
    const float* b2l   = (const float*)d_in[11];
    const float* W2r   = (const float*)d_in[12];
    const float* b2r   = (const float*)d_in[13];
    const float* att2  = (const float*)d_in[14];
    const float* bias2 = (const float*)d_in[15];

    const int N    = in_sizes[0] / D_IN;   // 20000
    const int E0   = in_sizes[1] / 2;      // 640000
    const int Etot = E0 + N;               // + self-loops
    const int* srcp = eidx;
    const int* dstp = eidx + E0;

    // ---- workspace bump allocator ----
    char*  ws  = (char*)d_ws;
    size_t off = 0;
    auto alloc = [&](size_t bytes) -> void* {
        off = (off + 255) & ~(size_t)255;
        void* p = ws + off;
        off += bytes;
        return p;
    };

    unsigned short* hbf    = (unsigned short*)alloc((size_t)N * D_IN * 2);
    unsigned short* w1lp   = (unsigned short*)alloc((size_t)D_IN * H1C1 * 2);
    unsigned short* w1rp   = (unsigned short*)alloc((size_t)D_IN * H1C1 * 2);
    unsigned short* w2lp   = (unsigned short*)alloc((size_t)H1C1 * H2C2 * 2);
    unsigned short* w2rp   = (unsigned short*)alloc((size_t)H1C1 * H2C2 * 2);
    float*          sums   = (float*)alloc(2 * D_IN * 4);
    float*          xl1    = (float*)alloc((size_t)N * H1C1 * 4);
    float*          xr1    = (float*)alloc((size_t)N * H1C1 * 4);
    float*          alpha1 = (float*)alloc((size_t)Etot * 4 * 4);
    unsigned int*   amax1  = (unsigned int*)alloc((size_t)N * 4 * 4);
    float*          denom1 = (float*)alloc((size_t)N * 4 * 4);
    float*          agg1   = (float*)alloc((size_t)N * H1C1 * 4);
    unsigned short* h1bf   = (unsigned short*)alloc((size_t)N * H1C1 * 2);
    unsigned int*   amax2  = (unsigned int*)alloc((size_t)N * 4);
    float*          denom2 = (float*)alloc((size_t)N * 4);

    // Aliased regions (stream-ordered; new producer strictly after last
    // consumer of the old use):
    float* xl2    = (float*)agg1;                 // agg1 dead after finalize1
    float* xr2    = xl2 + (size_t)N * H2C2;       // fits: N*256*4 == 2 * N*128*4
    float* alpha2 = (float*)hbf;                  // hbf dead after gemm1 (5.1MB >= 2.7MB)
    float* agg2   = (float*)alpha1;               // alpha1 dead after edge_agg1 (10.6MB >= 10.2MB)

    // ---- zero accumulators (every call; harness does not re-poison) ----
    hipMemsetAsync(sums,   0, 2 * D_IN * 4, stream);
    hipMemsetAsync(amax1,  0, (size_t)N * 4 * 4, stream);
    hipMemsetAsync(denom1, 0, (size_t)N * 4 * 4, stream);
    hipMemsetAsync(agg1,   0, (size_t)N * H1C1 * 4, stream);
    hipMemsetAsync(amax2,  0, (size_t)N * 4, stream);
    hipMemsetAsync(denom2, 0, (size_t)N * 4, stream);

    // ---- BatchNorm + bf16 cast ----
    bn_stats_kernel<<<512, 128, 0, stream>>>(x, sums, N);
    bn_apply_kernel<<<ceil_div((long long)N * D_IN, 256), 256, 0, stream>>>(
        x, sums, gamma, beta, hbf, N * D_IN, 1.0f / (float)N);

    // ---- pack weights into WMMA B-fragment layout (fused f32->bf16) ----
    {
        int t1 = (H1C1 >> 4) * (D_IN >> 5) * 32;   // 2048
        int t2 = (H2C2 >> 4) * (H1C1 >> 5) * 32;   // 2048
        pack_b_kernel<<<ceil_div(t1, 256), 256, 0, stream>>>(W1l, w1lp, H1C1, D_IN);
        pack_b_kernel<<<ceil_div(t1, 256), 256, 0, stream>>>(W1r, w1rp, H1C1, D_IN);
        pack_b_kernel<<<ceil_div(t2, 256), 256, 0, stream>>>(W2l, w2lp, H2C2, H1C1);
        pack_b_kernel<<<ceil_div(t2, 256), 256, 0, stream>>>(W2r, w2rp, H2C2, H1C1);
    }

    // ---- layer 1 projections (WMMA) : M=N, N=256, K=128 ----
    {
        int waves  = (N >> 4) * (H1C1 >> 5);      // 2 tiles per wave
        int blocks = ceil_div(waves, 4);          // 4 waves / 128-thread block
        wmma_gemm_bias_kernel<D_IN><<<blocks, 128, 0, stream>>>(hbf, w1lp, b1l, xl1, N, H1C1);
        wmma_gemm_bias_kernel<D_IN><<<blocks, 128, 0, stream>>>(hbf, w1rp, b1r, xr1, N, H1C1);
    }

    // ---- layer 1 edge softmax + aggregation ----
    {
        int eb = ceil_div(Etot, 8);               // 8 waves / 256-thread block
        edge_alpha_kernel<4, 64><<<eb, 256, 0, stream>>>(xl1, xr1, att1, srcp, dstp,
                                                         E0, Etot, alpha1, amax1);
        edge_exp_kernel<4><<<ceil_div((long long)Etot * 4, 256), 256, 0, stream>>>(
            alpha1, amax1, denom1, dstp, E0, Etot);
        edge_agg_kernel<4, 64><<<eb, 256, 0, stream>>>(xl1, alpha1, denom1, srcp, dstp,
                                                       E0, Etot, agg1);
    }

    // ---- + bias1, leaky_relu(0.01), cast to bf16 for layer-2 GEMM ----
    bias_lrelu_to_bf16_kernel<<<ceil_div((long long)N * H1C1, 256), 256, 0, stream>>>(
        agg1, bias1, h1bf, N * H1C1, H1C1);

    // agg2 aliases alpha1 — zero it now (alpha1 fully consumed above)
    hipMemsetAsync(agg2, 0, (size_t)N * H2C2 * 4, stream);

    // ---- layer 2 projections (WMMA) : M=N, N=128, K=256 ----
    {
        int waves  = (N >> 4) * (H2C2 >> 5);
        int blocks = ceil_div(waves, 4);
        wmma_gemm_bias_kernel<H1C1><<<blocks, 128, 0, stream>>>(h1bf, w2lp, b2l, xl2, N, H2C2);
        wmma_gemm_bias_kernel<H1C1><<<blocks, 128, 0, stream>>>(h1bf, w2rp, b2r, xr2, N, H2C2);
    }

    // ---- layer 2 edge softmax + aggregation ----
    {
        int eb = ceil_div(Etot, 8);
        edge_alpha_kernel<1, 128><<<eb, 256, 0, stream>>>(xl2, xr2, att2, srcp, dstp,
                                                          E0, Etot, alpha2, amax2);
        edge_exp_kernel<1><<<ceil_div(Etot, 256), 256, 0, stream>>>(
            alpha2, amax2, denom2, dstp, E0, Etot);
        edge_agg_kernel<1, 128><<<eb, 256, 0, stream>>>(xl2, alpha2, denom2, srcp, dstp,
                                                        E0, Etot, agg2);
    }

    // ---- final: out = agg2 + bias2 (f32) ----
    add_bias_out_kernel<<<ceil_div((long long)N * H2C2, 256), 256, 0, stream>>>(
        agg2, bias2, (float*)d_out, N * H2C2, H2C2);
}
